// VisionAttention_65936337928634
// MI455X (gfx1250) — matrix operand order repaired and checked by reading, not compiled, not run on hardware
//
#include <hip/hip_runtime.h>
#include <hip/hip_bf16.h>
#include <math.h>

// ---------------------------------------------------------------- constants
#define Bn   4
#define Sn   2048
#define Dn   1024
#define Hn   16
#define HDn  64
#define EPSf 1e-6f

typedef _Float16 half_t;
typedef __attribute__((ext_vector_type(16))) _Float16 v16h;
typedef __attribute__((ext_vector_type(8)))  _Float16 v8h;
typedef __attribute__((ext_vector_type(8)))  float    v8f;

__device__ __forceinline__ v8f vzero() {
    v8f v;
#pragma unroll
    for (int i = 0; i < 8; ++i) v[i] = 0.0f;
    return v;
}

// A-fragment (16x32 f16, MxK): lane<16 holds K {koff..koff+7, koff+16..koff+23}
// with koff = 0; lanes 16-31 the same rows with koff = 8.
__device__ __forceinline__ v16h load_a_frag(const half_t* p, int koff) {
    union { v16h v; half_t h[16]; } u;
#pragma unroll
    for (int i = 0; i < 8; ++i) {
        u.h[i]     = p[koff + i];
        u.h[i + 8] = p[koff + 16 + i];
    }
    return u.v;
}

// B-fragment (32x16 f16, KxN): lane holds column (lane&15), K {o..o+7, o+16..o+23}
// with o = kboff/2 (kboff = 0 for lanes 0-15, 16 for lanes 16-31).
__device__ __forceinline__ v16h load_b_frag(const half_t* p, int kboff) {
    union { v16h v; half_t h[16]; } u;
#pragma unroll
    for (int i = 0; i < 8; ++i) {
        u.h[i]     = p[(kboff >> 1) + i];
        u.h[i + 8] = p[(kboff >> 1) + 16 + i];
    }
    return u.v;
}

__device__ __forceinline__ v8f wmma16(v16h a, v16h b, v8f c) {
    return __builtin_amdgcn_wmma_f32_16x16x32_f16(false, a, false, b,
                                                  (short)0, c, false, false);
}

// Async DMA: 16 bytes global -> LDS, tracked by ASYNCcnt (CDNA5 path).
// GVS addressing: mem = SGPR64 + VGPR32 offset ; VDST = LDS byte address.
__device__ __forceinline__ void async_b128(void* lds, const void* gbase,
                                           uint32_t voff) {
    uint32_t l = (uint32_t)(uintptr_t)lds;   // low 32 bits of LDS aperture = LDS offset
    asm volatile("global_load_async_to_lds_b128 %0, %1, %2"
                 :: "v"(l), "v"(voff), "s"(gbase)
                 : "memory");
}
__device__ __forceinline__ void wait_async0() {
    asm volatile("s_wait_asynccnt 0x0" ::: "memory");
}

// =====================================================================
// Kernel 1: fused QKV projection (f32 in, f16 WMMA) + RMSnorm + RoPE-2D
// grid = (M/64, H) ; block = 256 (8 waves). One block: 64 tokens, 1 head,
// all of q,k,v for that head (3 GEMMs of 64x64 sharing the X tile).
// Wave pair (wave>>1) owns one 16-row M strip; B-fragments are software
// pipelined in registers so every WMMA waits only on DScnt<=2.
// =====================================================================
__global__ __launch_bounds__(256) void qkv_norm_rope_kernel(
    const float* __restrict__ X,         // (B*S, D)
    const float* __restrict__ wq,        // (H*HD, D)
    const float* __restrict__ wk,
    const float* __restrict__ wv,
    const float* __restrict__ q_scale,   // (HD,)
    const float* __restrict__ k_scale,   // (HD,)
    const float* __restrict__ cosp,      // (B,S,HD)
    const float* __restrict__ sinp,
    half_t* __restrict__ Qw,             // [B][H][S][HD]
    half_t* __restrict__ Kw,             // [B][H][S][HD]
    half_t* __restrict__ Vt)             // [B][H][HD][S]
{
    __shared__ __align__(16) char smem[49152];
    half_t* Xs  = (half_t*)smem;                 // [64][72]
    half_t* Wsm = Xs + 64 * 72;                  // [3][64][72]
    float*  Acc = (float*)smem;                  // [3][64][64] (repurposed)

    const int tid  = threadIdx.x;
    const int wave = tid >> 5;
    const int lane = tid & 31;
    const int l16  = lane & 15;
    const int koff  = (lane & 16) ? 8 : 0;
    const int kboff = (lane & 16) ? 16 : 0;

    const int mbase = blockIdx.x * 64;           // token tile
    const int h     = blockIdx.y;
    const int hbase = h * 64;                    // rows of wq/wk/wv for head h

    const int mi   = wave >> 1;                  // M strip owned by wave pair
    const int tsel = (wave & 1) * 6;             // 6 of 12 (o,ni) tiles

    // LDS row base (in half_t units) of the B tile for pipeline step st (0..11)
#define QKV_BROW(st)                                                           \
    ((((tsel + (st) % 6) >> 2) * 64 + ((tsel + (st) % 6) & 3) * 16 + l16) * 72 \
     + ((st) >= 6 ? 32 : 0))

    v8f acc[6];
#pragma unroll
    for (int i = 0; i < 6; ++i) acc[i] = vzero();

    for (int kc = 0; kc < Dn / 64; ++kc) {
        const int kb = kc * 64;
        __syncthreads();
        // stage X tile + 3 weight tiles, f32 -> f16 (4096 float4 groups)
        for (int idx = tid; idx < 4096; idx += 256) {
            const int which = idx >> 10;         // 0=X, 1=wq, 2=wk, 3=wv
            const int rem   = idx & 1023;
            const int row   = rem >> 4;
            const int c4    = (rem & 15) << 2;
            const float* src;
            if (which == 0)      src = X  + (size_t)(mbase + row) * Dn + kb + c4;
            else if (which == 1) src = wq + (size_t)(hbase + row) * Dn + kb + c4;
            else if (which == 2) src = wk + (size_t)(hbase + row) * Dn + kb + c4;
            else                 src = wv + (size_t)(hbase + row) * Dn + kb + c4;
            const float4 f = *(const float4*)src;
            half_t* dst = (which == 0) ? &Xs[row * 72 + c4]
                                       : &Wsm[((which - 1) * 64 + row) * 72 + c4];
            dst[0] = (half_t)f.x; dst[1] = (half_t)f.y;
            dst[2] = (half_t)f.z; dst[3] = (half_t)f.w;
        }
        __syncthreads();

        // flat 12-step WMMA pipeline: prefetch B(st+1) before WMMA(st)
        const int xrow = (mi * 16 + l16) * 72;
        const v16h a0 = load_a_frag(&Xs[xrow + 0],  koff);
        const v16h a1 = load_a_frag(&Xs[xrow + 32], koff);
        v16h bn = load_b_frag(&Wsm[QKV_BROW(0)], kboff);
#pragma unroll
        for (int st = 0; st < 12; ++st) {
            const v16h bc = bn;
            if (st < 11) bn = load_b_frag(&Wsm[QKV_BROW(st + 1)], kboff);
            acc[st % 6] = wmma16(st < 6 ? a0 : a1, bc, acc[st % 6]);
        }
    }
#undef QKV_BROW

    // spill accumulators to LDS (C layout: vgpr r -> row r + 8*(lane>=16))
    __syncthreads();
    const int rowb = (lane & 16) ? 8 : 0;
#pragma unroll
    for (int i = 0; i < 6; ++i) {
        const int t2 = tsel + i;
        const int o  = t2 >> 2;
        const int ni = t2 & 3;
#pragma unroll
        for (int rr = 0; rr < 8; ++rr)
            Acc[(o * 64 + mi * 16 + rowb + rr) * 64 + ni * 16 + l16] = acc[i][rr];
    }
    __syncthreads();

    // RMS norm + RoPE-2D epilogue: one thread per (output, token-row)
    if (tid < 192) {
        const int o   = tid >> 6;                // 0=q,1=k,2=v
        const int row = tid & 63;
        const float* xr = &Acc[(o * 64 + row) * 64];

        float ss = 0.0f;
#pragma unroll
        for (int j = 0; j < HDn; ++j) { const float v = xr[j]; ss += v * v; }
        const float rms = rsqrtf(ss * (1.0f / HDn) + EPSf);

        const int g = mbase + row;               // global token
        const int b = g >> 11;                   // / S
        const int s = g & (Sn - 1);
        const float* cp = cosp + (size_t)g * HDn;
        const float* sp = sinp + (size_t)g * HDn;

        for (int j = 0; j < HDn; ++j) {
            float sc_j = (o == 0) ? q_scale[j] : (o == 1) ? k_scale[j] : 1.0f;
            float xv = xr[j] * rms * sc_j;
            float yv;
            if (o < 2) {
                const int j2 = j & 31;           // PER_DIM=32, HALF=16
                const int rj = (j2 < 16) ? j + 16 : j - 16;
                float sc_r = (o == 0) ? q_scale[rj] : k_scale[rj];
                float rv = xr[rj] * rms * sc_r;
                rv = (j2 < 16) ? -rv : rv;
                yv = xv * cp[j] + rv * sp[j];
            } else {
                yv = xv;
            }
            const size_t bh = (size_t)(b * Hn + h);
            if (o == 0)      Qw[(bh * Sn + s) * HDn + j] = (half_t)yv;
            else if (o == 1) Kw[(bh * Sn + s) * HDn + j] = (half_t)yv;
            else             Vt[(bh * HDn + j) * Sn + s] = (half_t)yv;
        }
    }
}

// =====================================================================
// Kernel 2: flash attention per (b,h). grid = (S/64, B*H), block = 128
// (4 waves); each wave owns a 16-query strip. K/V tiles are DMA'd into
// double-buffered LDS with global_load_async_to_lds_b128 (ASYNCcnt);
// chunk k+1's DMA overlaps chunk k's WMMAs. B-frags software-pipelined.
// =====================================================================
__global__ __launch_bounds__(128) void flash_attn_kernel(
    const half_t* __restrict__ Qw,       // [B][H][S][HD]
    const half_t* __restrict__ Kw,
    const half_t* __restrict__ Vt,       // [B][H][HD][S]
    half_t* __restrict__ AO)             // [B][S][H*HD]
{
    __shared__ __align__(16) half_t Ks[2][64][72];   // 18432 B
    __shared__ __align__(16) half_t Vs[2][64][72];   // 18432 B
    __shared__ __align__(16) half_t Ps[4][16][72];   //  9216 B

    const int bh    = blockIdx.y;
    const int qbase = blockIdx.x * 64;
    const int wave  = threadIdx.x >> 5;
    const int lane  = threadIdx.x & 31;
    const int l16   = lane & 15;
    const int koff  = (lane & 16) ? 8 : 0;
    const int kboff = (lane & 16) ? 16 : 0;
    const int rowb  = (lane & 16) ? 8 : 0;

    const half_t* Qp = Qw + (size_t)bh * Sn * HDn;
    const half_t* Kp = Kw + (size_t)bh * Sn * HDn;
    const half_t* Vp = Vt + (size_t)bh * HDn * Sn;

    const int qr = qbase + wave * 16;
    const v16h aq0 = load_a_frag(Qp + (size_t)(qr + l16) * HDn + 0,  koff);
    const v16h aq1 = load_a_frag(Qp + (size_t)(qr + l16) * HDn + 32, koff);

    v8f oa[4];
#pragma unroll
    for (int i = 0; i < 4; ++i) oa[i] = vzero();
    float mS[8], lS[8];
#pragma unroll
    for (int r = 0; r < 8; ++r) { mS[r] = -3.0e38f; lS[r] = 0.0f; }

    // ---- async stage of one 64-key chunk: wave stages 16 K rows + 16 V rows
#define FLASH_STAGE(bufi, kbv)                                                 \
    do {                                                                       \
        const half_t* kbase = Kp + (size_t)(kbv) * HDn; /* contiguous 8 KB */  \
        const half_t* vbase = Vp + (kbv);               /* rows stride S */    \
        _Pragma("unroll")                                                      \
        for (int i_ = 0; i_ < 4; ++i_) {                                       \
            const int chunk_ = lane + 32 * i_;          /* 0..127 */           \
            const int r16_   = chunk_ >> 3;                                    \
            const int c_     = chunk_ & 7;                                     \
            const int row_   = wave * 16 + r16_;        /* 0..63 */            \
            async_b128(&Ks[bufi][row_][c_ * 8], kbase,                         \
                       (uint32_t)(row_ * (HDn * 2) + c_ * 16));                \
            async_b128(&Vs[bufi][row_][c_ * 8], vbase,                         \
                       (uint32_t)(row_ * (Sn * 2) + c_ * 16));                 \
        }                                                                      \
    } while (0)

    FLASH_STAGE(0, 0);

    for (int kc = 0; kc < Sn / 64; ++kc) {
        const int kb  = kc * 64;
        const int cur = kc & 1;

        wait_async0();          // buffer `cur` DMA complete (only batch in flight)
        __syncthreads();        // ...and every wave is done reading buffer cur^1
        if (kc + 1 < Sn / 64)
            FLASH_STAGE(cur ^ 1, kb + 64);   // DMA overlaps the WMMAs below

        // ---- scores: 16x64 strip, 8-step pipelined WMMAs (B-frags from LDS)
        v8f sc[4];
        {
            v16h bn = load_b_frag(&Ks[cur][l16][0], kboff);
#pragma unroll
            for (int st = 0; st < 8; ++st) {
                const int nt  = st >> 1;
                const int ks2 = st & 1;
                const v16h bc = bn;
                if (st < 7) {
                    const int s2 = st + 1;
                    bn = load_b_frag(&Ks[cur][(s2 >> 1) * 16 + l16][(s2 & 1) * 32],
                                     kboff);
                }
                const v8f cin = ks2 ? sc[nt] : vzero();
                sc[nt] = wmma16(ks2 ? aq1 : aq0, bc, cin);
            }
        }

        // ---- online softmax: row max / rescale / exp / row sum
        float alpha[8];
#pragma unroll
        for (int r = 0; r < 8; ++r) {
            float v = fmaxf(fmaxf(sc[0][r], sc[1][r]), fmaxf(sc[2][r], sc[3][r]));
#pragma unroll
            for (int m = 1; m < 16; m <<= 1) v = fmaxf(v, __shfl_xor(v, m, 32));
            const float nm = fmaxf(mS[r], v);
            alpha[r] = __expf(mS[r] - nm);
            mS[r] = nm;
        }
#pragma unroll
        for (int r = 0; r < 8; ++r) {
            float rs = 0.0f;
#pragma unroll
            for (int nt = 0; nt < 4; ++nt) {
                const float p = __expf(sc[nt][r] - mS[r]);
                sc[nt][r] = p;
                rs += p;
            }
#pragma unroll
            for (int m = 1; m < 16; m <<= 1) rs += __shfl_xor(rs, m, 32);
            lS[r] = lS[r] * alpha[r] + rs;
#pragma unroll
            for (int nt = 0; nt < 4; ++nt) oa[nt][r] = oa[nt][r] * alpha[r];
        }

        // ---- P (C layout) -> LDS f16 -> A fragments (wave-local, no barrier)
#pragma unroll
        for (int nt = 0; nt < 4; ++nt)
#pragma unroll
            for (int r = 0; r < 8; ++r)
                Ps[wave][rowb + r][nt * 16 + l16] = (half_t)sc[nt][r];
        asm volatile("s_wait_dscnt 0x0" ::: "memory");
        const v16h ap0 = load_a_frag(&Ps[wave][l16][0],  koff);
        const v16h ap1 = load_a_frag(&Ps[wave][l16][32], koff);

        // ---- O += P @ V : 8-step pipelined (V rows = head-dim, from LDS)
        {
            v16h bn = load_b_frag(&Vs[cur][l16][0], kboff);
#pragma unroll
            for (int st = 0; st < 8; ++st) {
                const int nt  = st >> 1;
                const int ks2 = st & 1;
                const v16h bc = bn;
                if (st < 7) {
                    const int s2 = st + 1;
                    bn = load_b_frag(&Vs[cur][(s2 >> 1) * 16 + l16][(s2 & 1) * 32],
                                     kboff);
                }
                oa[nt] = wmma16(ks2 ? ap1 : ap0, bc, oa[nt]);
            }
        }
        __syncthreads();        // all waves done with buffer cur before reuse
    }

    // ---- finalize: O /= l ; write f16 to [B][S][H*HD]
    const int b = bh >> 4;                        // / H
    const int h = bh & (Hn - 1);
#pragma unroll
    for (int r = 0; r < 8; ++r) {
        const float inv = 1.0f / lS[r];
        const int srow = qbase + wave * 16 + rowb + r;
#pragma unroll
        for (int nt = 0; nt < 4; ++nt)
            AO[((size_t)b * Sn + srow) * (Hn * HDn) + h * HDn + nt * 16 + l16] =
                (half_t)(oa[nt][r] * inv);
    }
#undef FLASH_STAGE
}

// =====================================================================
// Kernel 3: output projection  out = AO(f16) @ wo^T(f32->f16), f32 out.
// grid = (M/64, N/64), block = 256 (8 waves x 2 tiles, both tiles share
// the wave's M strip). A-tile async DMA'd to LDS; B-tile f32->f16 VALU.
// =====================================================================
__global__ __launch_bounds__(256) void out_proj_kernel(
    const half_t* __restrict__ AO,       // (B*S, 1024) f16
    const float*  __restrict__ wo,       // (D, H*HD) f32
    float* __restrict__ out)             // (B*S, D) f32
{
    __shared__ __align__(16) half_t As[64][72];
    __shared__ __align__(16) half_t Bs[64][72];

    const int tid  = threadIdx.x;
    const int wave = tid >> 5;
    const int lane = tid & 31;
    const int l16  = lane & 15;
    const int koff  = (lane & 16) ? 8 : 0;
    const int kboff = (lane & 16) ? 16 : 0;
    const int rowb  = (lane & 16) ? 8 : 0;

    const int mbase = blockIdx.x * 64;
    const int nbase = blockIdx.y * 64;
    const int mi    = wave >> 1;                 // both tiles share this strip

    v8f acc[2];
    acc[0] = vzero(); acc[1] = vzero();

    for (int kc = 0; kc < (Hn * HDn) / 64; ++kc) {
        const int kb = kc * 64;
        __syncthreads();
        // stage A via async DMA: 512 x 16B chunks, rows stride 2048 B
        {
            const half_t* abase = AO + (size_t)mbase * (Hn * HDn) + kb;
            for (int idx = tid; idx < 512; idx += 256) {
                const int row = idx >> 3;
                const int c   = idx & 7;
                async_b128(&As[row][c * 8], abase,
                           (uint32_t)(row * (Hn * HDn * 2) + c * 16));
            }
        }
        // stage B (f32 -> f16): 1024 float4 groups
        for (int idx = tid; idx < 1024; idx += 256) {
            const int row = idx >> 4;
            const int c4  = (idx & 15) << 2;
            const float4 f = *(const float4*)(wo + (size_t)(nbase + row) * (Hn * HDn) + kb + c4);
            Bs[row][c4 + 0] = (half_t)f.x; Bs[row][c4 + 1] = (half_t)f.y;
            Bs[row][c4 + 2] = (half_t)f.z; Bs[row][c4 + 3] = (half_t)f.w;
        }
        wait_async0();
        __syncthreads();

        // flat 4-step pipeline: (ks,i), A shared per ks, B prefetched
        {
            const v16h a0 = load_a_frag(&As[mi * 16 + l16][0],  koff);
            const v16h a1 = load_a_frag(&As[mi * 16 + l16][32], koff);
            v16h bn = load_b_frag(&Bs[((wave * 2) & 3) * 16 + l16][0], kboff);
#pragma unroll
            for (int st = 0; st < 4; ++st) {
                const int ks2 = st >> 1;         // 0,0,1,1
                const int i   = st & 1;
                const v16h bc = bn;
                if (st < 3) {
                    const int s2 = st + 1;
                    const int ni2 = ((wave * 2 + (s2 & 1)) & 3);
                    bn = load_b_frag(&Bs[ni2 * 16 + l16][(s2 >> 1) * 32], kboff);
                }
                acc[i] = wmma16(ks2 ? a1 : a0, bc, acc[i]);
            }
        }
    }

#pragma unroll
    for (int i = 0; i < 2; ++i) {
        const int ni = (wave * 2 + i) & 3;
#pragma unroll
        for (int r = 0; r < 8; ++r)
            out[(size_t)(mbase + mi * 16 + rowb + r) * Dn + nbase + ni * 16 + l16] =
                acc[i][r];
    }
}

// ---------------------------------------------------------------- launch
extern "C" void kernel_launch(void* const* d_in, const int* in_sizes, int n_in,
                              void* d_out, int out_size, void* d_ws, size_t ws_size,
                              hipStream_t stream) {
    (void)in_sizes; (void)n_in; (void)out_size; (void)ws_size;

    const float* X    = (const float*)d_in[0];   // hidden_states
    const float* cosp = (const float*)d_in[1];
    const float* sinp = (const float*)d_in[2];
    // d_in[3] = position_ids (unused by the reference computation)
    const float* wq   = (const float*)d_in[4];
    const float* wk   = (const float*)d_in[5];
    const float* wv   = (const float*)d_in[6];
    const float* wo   = (const float*)d_in[7];
    const float* qs   = (const float*)d_in[8];
    const float* ks   = (const float*)d_in[9];

    const size_t QSZ = (size_t)Bn * Hn * Sn * HDn;   // 8M f16 elements
    half_t* Qw = (half_t*)d_ws;
    half_t* Kw = Qw + QSZ;
    half_t* Vt = Kw + QSZ;
    half_t* AO = Vt + QSZ;                           // 64 MB total in d_ws

    qkv_norm_rope_kernel<<<dim3((Bn * Sn) / 64, Hn), 256, 0, stream>>>(
        X, wq, wk, wv, qs, ks, cosp, sinp, Qw, Kw, Vt);

    flash_attn_kernel<<<dim3(Sn / 64, Bn * Hn), 128, 0, stream>>>(
        Qw, Kw, Vt, AO);

    out_proj_kernel<<<dim3((Bn * Sn) / 64, Dn / 64), 256, 0, stream>>>(
        AO, wo, (float*)d_out);
}
